// QLoRALinear_34454227649197
// MI455X (gfx1250) — compile-verified, matrix operand order checked by reading, not run on hardware
//
#include <hip/hip_runtime.h>
#include <stddef.h>

// Problem constants (match the JAX reference)
#define TOKN  16384
#define INF   4096
#define OUTF  4096
#define RLORA 16
#define GSZ   64
#define NGRP  (INF / GSZ)   // 64 groups per row

// Tiling
#define BM  128
#define BN  128
#define BK  64              // one quant group per K-chunk
#define LDH 72              // padded LDS row stride in halves (144 B)

typedef __attribute__((ext_vector_type(8)))  _Float16 v8h;
typedef __attribute__((ext_vector_type(16))) _Float16 v16h;
typedef __attribute__((ext_vector_type(8)))  float    v8f;
typedef __attribute__((ext_vector_type(4)))  float    v4f;

union AF { v16h v; v8h h[2]; };

// ---------------------------------------------------------------------------
// Stage 1: T[tok, r] = sum_k x[tok,k] * lora_A[r,k]    (f32, tiny: ~2 GFLOP)
// One block = 16 tokens x 16 ranks; x[tok,k] is a broadcast load across the
// 16 lanes that share a token, lora_A rows live in L2.
// ---------------------------------------------------------------------------
__global__ __launch_bounds__(256) void lora_stage1_kernel(
    const _Float16* __restrict__ x,
    const float*    __restrict__ loraA,
    float*          __restrict__ T)
{
    const int tok = blockIdx.x * 16 + (threadIdx.x >> 4);
    const int r   = threadIdx.x & 15;
    const _Float16* xr = x     + (size_t)tok * INF;
    const float*    ar = loraA + (size_t)r   * INF;

    float acc = 0.0f;
    for (int k = 0; k < INF; k += 8) {
        v8h xv = *(const v8h*)(xr + k);
        v4f a0 = *(const v4f*)(ar + k);
        v4f a1 = *(const v4f*)(ar + k + 4);
        acc += (float)xv[0] * a0[0] + (float)xv[1] * a0[1]
             + (float)xv[2] * a0[2] + (float)xv[3] * a0[3]
             + (float)xv[4] * a1[0] + (float)xv[5] * a1[1]
             + (float)xv[6] * a1[2] + (float)xv[7] * a1[3];
    }
    T[(size_t)tok * RLORA + r] = acc;
}

// Dequantize 8 int4 codes (stored as int32) with one group scale -> v8h
__device__ inline v8h dq8(int4 a, int4 b, float s) {
    v8h w;
    w[0] = (_Float16)(s * (float)(a.x - 8));
    w[1] = (_Float16)(s * (float)(a.y - 8));
    w[2] = (_Float16)(s * (float)(a.z - 8));
    w[3] = (_Float16)(s * (float)(a.w - 8));
    w[4] = (_Float16)(s * (float)(b.x - 8));
    w[5] = (_Float16)(s * (float)(b.y - 8));
    w[6] = (_Float16)(s * (float)(b.z - 8));
    w[7] = (_Float16)(s * (float)(b.w - 8));
    return w;
}

// ---------------------------------------------------------------------------
// Main kernel: C[m,n] = sum_k x[m,k] * w[n,k]  (w dequantized on the fly),
// plus fused LoRA epilogue C += T[m,:] . lora_B[n,:].
// Block: 128x128 tile, 8 waves; wave: 32x64 = 2x4 accumulators of
// v_wmma_f32_16x16x32_f16.
// ---------------------------------------------------------------------------
__global__ __launch_bounds__(256) void qlora_gemm_kernel(
    const _Float16* __restrict__ x,
    const int*      __restrict__ qw,
    const float*    __restrict__ scales,
    const float*    __restrict__ loraB,
    const float*    __restrict__ T,
    _Float16*       __restrict__ out)
{
    __shared__ _Float16 As[BM * LDH];   // x tile, [m][k], padded
    __shared__ _Float16 Ws[BN * LDH];   // dequantized weight tile, [n][k], padded

    const int tid  = threadIdx.x;
    const int lane = tid & 31;
    const int wave = tid >> 5;
    const int l16  = lane & 15;
    const int hl   = lane >> 4;         // lane half: selects K sub-ranges

    const int m0 = blockIdx.x * BM;     // token base
    const int n0 = blockIdx.y * BN;     // output-feature base

    const int mw = (wave >> 1) * 32;    // wave offsets inside the block tile
    const int nw = (wave & 1) * 64;

    v8f acc[2][4];
    #pragma unroll
    for (int i = 0; i < 2; ++i)
        #pragma unroll
        for (int j = 0; j < 4; ++j)
            acc[i][j] = (v8f){};

    // cooperative-load assignment: 256 threads cover 128 rows x 2 half-chunks
    const int lr = tid >> 1;            // 0..127 row
    const int lc = (tid & 1) * 32;      // 0 or 32 (k offset within chunk)

    const _Float16* xg = x  + (size_t)(m0 + lr) * INF + lc;
    const int*      qg = qw + (size_t)(n0 + lr) * INF + lc;
    const float*    sg = scales + (size_t)(n0 + lr) * NGRP;

    for (int kk = 0; kk < INF; kk += BK) {
        // ---- global loads into registers ----
        const v8h* xp = (const v8h*)(xg + kk);
        v8h x0 = xp[0], x1 = xp[1], x2 = xp[2], x3 = xp[3];

        const int4* qp = (const int4*)(qg + kk);
        int4 q0 = qp[0], q1 = qp[1], q2 = qp[2], q3 = qp[3];
        int4 q4 = qp[4], q5 = qp[5], q6 = qp[6], q7 = qp[7];
        const float s = sg[kk >> 6];    // BK==GSZ: one scale per chunk per row

        // prefetch next K-chunk (global_prefetch_b8); speculative, OOB dropped
        __builtin_prefetch((const void*)(xg + kk + BK), 0, 0);
        __builtin_prefetch((const void*)(qg + kk + BK), 0, 0);

        // ---- stage into LDS ----
        {
            v8h* ax = (v8h*)&As[lr * LDH + lc];
            ax[0] = x0; ax[1] = x1; ax[2] = x2; ax[3] = x3;
            v8h* aw = (v8h*)&Ws[lr * LDH + lc];
            aw[0] = dq8(q0, q1, s);
            aw[1] = dq8(q2, q3, s);
            aw[2] = dq8(q4, q5, s);
            aw[3] = dq8(q6, q7, s);
        }
        __syncthreads();

        // ---- WMMA over the two 32-wide K sub-chunks ----
        #pragma unroll
        for (int kc = 0; kc < BK; kc += 32) {
            AF a[2], b[4];
            // 16-bit A layout: lanes 0-15 hold K {0..7,16..23}, lanes 16-31
            // hold K {8..15,24..31} of their row.
            #pragma unroll
            for (int i = 0; i < 2; ++i) {
                const int row = mw + i * 16 + l16;
                a[i].h[0] = *(const v8h*)&As[row * LDH + kc + hl * 8];
                a[i].h[1] = *(const v8h*)&As[row * LDH + kc + 16 + hl * 8];
            }
            #pragma unroll
            for (int j = 0; j < 4; ++j) {
                const int row = nw + j * 16 + l16;
                b[j].h[0] = *(const v8h*)&Ws[row * LDH + kc + hl * 8];
                b[j].h[1] = *(const v8h*)&Ws[row * LDH + kc + 16 + hl * 8];
            }
            #pragma unroll
            for (int i = 0; i < 2; ++i)
                #pragma unroll
                for (int j = 0; j < 4; ++j)
                    acc[i][j] = __builtin_amdgcn_wmma_f32_16x16x32_f16(
                        false, a[i].v, false, b[j].v,
                        (short)0, acc[i][j], false, false);
        }
        __syncthreads();
    }

    // ---- epilogue: fuse rank-16 LoRA and store f16 (non-temporal) ----
    // C/D layout: lanes 0-15 -> N=l16, VGPR r -> M=r; lanes 16-31 -> M=r+8.
    #pragma unroll
    for (int j = 0; j < 4; ++j) {
        const int n = n0 + nw + j * 16 + l16;
        const float* bp = loraB + (size_t)n * RLORA;
        v4f b0 = *(const v4f*)(bp + 0);
        v4f b1 = *(const v4f*)(bp + 4);
        v4f b2 = *(const v4f*)(bp + 8);
        v4f b3 = *(const v4f*)(bp + 12);
        #pragma unroll
        for (int i = 0; i < 2; ++i) {
            const int mbase = m0 + mw + i * 16 + hl * 8;
            #pragma unroll
            for (int rr = 0; rr < 8; ++rr) {
                const int m = mbase + rr;
                const float* tp = T + (size_t)m * RLORA;
                v4f t0 = *(const v4f*)(tp + 0);
                v4f t1 = *(const v4f*)(tp + 4);
                v4f t2 = *(const v4f*)(tp + 8);
                v4f t3 = *(const v4f*)(tp + 12);
                float v = acc[i][j][rr]
                    + t0[0]*b0[0] + t0[1]*b0[1] + t0[2]*b0[2] + t0[3]*b0[3]
                    + t1[0]*b1[0] + t1[1]*b1[1] + t1[2]*b1[2] + t1[3]*b1[3]
                    + t2[0]*b2[0] + t2[1]*b2[1] + t2[2]*b2[2] + t2[3]*b2[3]
                    + t3[0]*b3[0] + t3[1]*b3[1] + t3[2]*b3[2] + t3[3]*b3[3];
                __builtin_nontemporal_store((_Float16)v,
                                            out + (size_t)m * OUTF + n);
            }
        }
    }
}

// ---------------------------------------------------------------------------
extern "C" void kernel_launch(void* const* d_in, const int* in_sizes, int n_in,
                              void* d_out, int out_size, void* d_ws, size_t ws_size,
                              hipStream_t stream) {
    (void)in_sizes; (void)n_in; (void)out_size; (void)ws_size;
    const _Float16* x      = (const _Float16*)d_in[0];
    const int*      qw     = (const int*)d_in[1];
    const float*    scales = (const float*)d_in[2];
    const float*    loraA  = (const float*)d_in[3];
    const float*    loraB  = (const float*)d_in[4];
    _Float16*       out    = (_Float16*)d_out;
    float*          T      = (float*)d_ws;   // TOKN * RLORA * 4 = 1 MB scratch

    lora_stage1_kernel<<<TOKN / 16, 256, 0, stream>>>(x, loraA, T);

    dim3 grid(TOKN / BM, OUTF / BN);
    qlora_gemm_kernel<<<grid, 256, 0, stream>>>(x, qw, scales, loraB, T, out);
}
